// FeatureExtractorWarped_13975823581674
// MI455X (gfx1250) — matrix operand order repaired and checked by reading, not compiled, not run on hardware
//
#include <hip/hip_runtime.h>
#include <hip/hip_bf16.h>
#include <math.h>

// ---------------------------------------------------------------------------
// FeatureExtractorWarped pipeline for MI455X (gfx1250, wave32, WMMA).
//  * convs = implicit GEMM via v_wmma_f32_16x16x32_f16 (f32 accumulate)
//  * weights pre-converted to f16 (rows padded to K%32==0) so the A-tile is a
//    raw contiguous copy -> staged with GLOBAL_LOAD_ASYNC_TO_LDS_B64 when the
//    toolchain exposes the builtin (guarded), else plain LDS stores
//  * activations kept in f16 end-to-end; splat accumulators stay f32
// ---------------------------------------------------------------------------

typedef _Float16 v16h __attribute__((ext_vector_type(16)));
typedef float    v8f  __attribute__((ext_vector_type(8)));
typedef int      v2i32 __attribute__((ext_vector_type(2)));

#define CONV_BLOCK 128   // 4 waves of 32
#define TILE_N     64    // pixels per block (16 per wave)
#define TILE_M     32    // output channels per block (2 WMMA M-tiles)

#define AS1 __attribute__((address_space(1)))
#define AS3 __attribute__((address_space(3)))

#if __has_builtin(__builtin_amdgcn_global_load_async_to_lds_b64)
#define USE_ASYNC_LDS 1
#else
#define USE_ASYNC_LDS 0
#endif

__device__ __forceinline__ void wait_async_lds()
{
#if __has_builtin(__builtin_amdgcn_s_wait_asynccnt)
    __builtin_amdgcn_s_wait_asynccnt(0);
#else
    asm volatile("s_wait_asynccnt 0x0" ::: "memory");
#endif
}

// Generic 3x3 conv, padding 1, arbitrary stride, NCHW, fused bias (+ SiLU).
// GEMM view: M = O (tile 32), N = Ho*Wo (tile 64), K = C*9 padded to Kp.
// Weights: f16, layout O x Kp with zero pad for k in [K, Kp).
__global__ __launch_bounds__(CONV_BLOCK)
void conv3x3_wmma_kernel(const _Float16* __restrict__ X,
                         const _Float16* __restrict__ Wt,
                         const float* __restrict__ Bias,
                         _Float16* __restrict__ Yh, float* __restrict__ Yf,
                         int C, int H, int W, int O, int stride,
                         int Ho, int Wo, int Kp, int act)
{
    __shared__ _Float16 As[TILE_M * 32];     // A[m][k]  (weights tile, 2 KB)
    __shared__ _Float16 Bs[32 * TILE_N];     // B[k][n]  (im2col tile, 4 KB)

    const int tid    = threadIdx.x;
    const int wave   = tid >> 5;
    const int lane   = tid & 31;
    const int coBase = blockIdx.y * TILE_M;
    const int nBase  = blockIdx.x * TILE_N;
    const int K      = C * 9;                // real reduction depth (<= Kp)
    const int nCh    = Kp >> 5;
    const int HWo    = Ho * Wo;

    v8f acc0 = {0.f,0.f,0.f,0.f,0.f,0.f,0.f,0.f};
    v8f acc1 = {0.f,0.f,0.f,0.f,0.f,0.f,0.f,0.f};

    for (int ch = 0; ch < nCh; ++ch) {
        const int kBase = ch << 5;

        // ---- stage A: 32x32 f16 weights (raw contiguous row segments)
#if USE_ASYNC_LDS
        #pragma unroll
        for (int it = 0; it < 2; ++it) {
            int idx = tid + it * CONV_BLOCK;      // 256 b64 copies
            int m   = idx >> 3;                   // 0..31
            int q   = idx & 7;                    // 8 x (4 f16) per row
            const _Float16* src = Wt + (size_t)(coBase + m) * Kp + kBase + 4 * q;
            __builtin_amdgcn_global_load_async_to_lds_b64(
                (AS1 v2i32*)src, (AS3 v2i32*)&As[(m << 5) + 4 * q], 0, 0);
        }
#else
        #pragma unroll
        for (int it = 0; it < (TILE_M * 32) / CONV_BLOCK; ++it) {
            int i = tid + it * CONV_BLOCK;
            int m = i >> 5;
            int k = i & 31;
            As[i] = Wt[(size_t)(coBase + m) * Kp + kBase + k];
        }
#endif
        if (kBase + 32 < Kp)
            __builtin_prefetch(Wt + (size_t)coBase * Kp + kBase + 32, 0, 1);

        // ---- stage B: 32 x 64 f16 im2col activations with zero padding
        #pragma unroll
        for (int it = 0; it < (32 * TILE_N) / CONV_BLOCK; ++it) {
            int i  = tid + it * CONV_BLOCK;
            int k  = i >> 6;               // TILE_N == 64
            int n  = i & (TILE_N - 1);
            int kk = kBase + k;
            _Float16 v = (_Float16)0.f;
            if (kk < K) {
                int ci = kk / 9;
                int t  = kk - ci * 9;
                int ky = t / 3;
                int kx = t - ky * 3;
                int p  = nBase + n;
                int oy = p / Wo;
                int ox = p - oy * Wo;
                int iy = oy * stride + ky - 1;
                int ix = ox * stride + kx - 1;
                if (iy >= 0 && iy < H && ix >= 0 && ix < W)
                    v = X[((size_t)ci * H + iy) * W + ix];
            }
            Bs[k * TILE_N + n] = v;
        }
#if USE_ASYNC_LDS
        wait_async_lds();
#endif
        __syncthreads();

        // ---- fragments: A 16x32 layout (lane 0-15: M, K 0-7/16-23;
        //      lane 16-31: same M, K 8-15/24-31), B 32x16 (lane%16 = N, same K split)
        v16h a0, a1, bfrag;
        {
            const int m  = lane & 15;
            const int kh = (lane >> 4) << 3;
            #pragma unroll
            for (int r = 0; r < 8; ++r) {
                int kk = (r < 4) ? (kh + 2 * r) : (16 + kh + 2 * (r - 4));
                a0[2 * r]     = As[(m << 5) + kk];
                a0[2 * r + 1] = As[(m << 5) + kk + 1];
                a1[2 * r]     = As[((m + 16) << 5) + kk];
                a1[2 * r + 1] = As[((m + 16) << 5) + kk + 1];
            }
        }
        {
            const int nOff = (wave << 4) + (lane & 15);
            const int kh = (lane >> 4) << 3;
            #pragma unroll
            for (int r = 0; r < 8; ++r) {
                int kk = (r < 4) ? (kh + 2 * r) : (16 + kh + 2 * (r - 4));
                bfrag[2 * r]     = Bs[kk * TILE_N + nOff];
                bfrag[2 * r + 1] = Bs[(kk + 1) * TILE_N + nOff];
            }
        }

        acc0 = __builtin_amdgcn_wmma_f32_16x16x32_f16(false, a0, false, bfrag,
                                                      (short)0, acc0, false, false);
        acc1 = __builtin_amdgcn_wmma_f32_16x16x32_f16(false, a1, false, bfrag,
                                                      (short)0, acc1, false, false);
        __syncthreads();
    }

    // ---- C/D layout: lane%16 = N, VGPR r -> M = r + 8*(lane/16)
    const int n  = nBase + (wave << 4) + (lane & 15);
    const int mr = ((lane >> 4) << 3);
    #pragma unroll
    for (int r = 0; r < 8; ++r) {
        int   m0 = coBase + r + mr;
        int   m1 = m0 + 16;
        float v0 = acc0[r] + Bias[m0];
        float v1 = acc1[r] + Bias[m1];
        if (act) {
            v0 = v0 / (1.f + __expf(-v0));     // SiLU
            v1 = v1 / (1.f + __expf(-v1));
        }
        if (Yf) {
            Yf[(size_t)m0 * HWo + n] = v0;
            Yf[(size_t)m1 * HWo + n] = v1;
        } else {
            Yh[(size_t)m0 * HWo + n] = (_Float16)v0;
            Yh[(size_t)m1 * HWo + n] = (_Float16)v1;
        }
    }
}

// ---------------------------------------------------------------------------
// fp32 -> f16 conversion kernels
// ---------------------------------------------------------------------------

// weights: O x K fp32 -> O x Kp f16, zero pad k in [K, Kp)
__global__ void cvt_weight_kernel(const float* __restrict__ src,
                                  _Float16* __restrict__ dst,
                                  int K, int Kp, int total)
{
    int i = blockIdx.x * blockDim.x + threadIdx.x;
    if (i >= total) return;
    int o = i / Kp;
    int k = i - o * Kp;
    dst[i] = (k < K) ? (_Float16)src[(size_t)o * K + k] : (_Float16)0.f;
}

__global__ void cvt_act_kernel(const float* __restrict__ src,
                               _Float16* __restrict__ dst, int n)
{
    int i = blockIdx.x * blockDim.x + threadIdx.x;
    if (i < n) dst[i] = (_Float16)src[i];
}

// ---------------------------------------------------------------------------
// Softmax splat (forward warp), metric == 1
// ---------------------------------------------------------------------------

__global__ void zero_kernel(float* __restrict__ p, int n)
{
    int i = blockIdx.x * blockDim.x + threadIdx.x;
    if (i < n) p[i] = 0.f;
}

__global__ void splat_kernel(const _Float16* __restrict__ feat,
                             const float* __restrict__ flow,
                             float* __restrict__ acc, int Hq, int Wq)
{
    const int p  = blockIdx.x * blockDim.x + threadIdx.x;  // source pixel
    const int c  = blockIdx.y;                             // channel 0..128
    const int HW = Hq * Wq;
    if (p >= HW) return;

    const float E  = 2.71828182845904523536f;   // exp(metric == 1)
    const float fx = (float)(p % Wq) + flow[p];
    const float fy = (float)(p / Wq) + flow[HW + p];
    const float val = (c < 128) ? (float)feat[(size_t)c * HW + p] * E : E;

    const float x0 = floorf(fx), y0 = floorf(fy);
    const int x0i = (int)x0, y0i = (int)y0;
    const float wx1 = fx - x0, wx0 = 1.f - wx1;
    const float wy1 = fy - y0, wy0 = 1.f - wy1;

    float* base = acc + (size_t)c * HW;
    #pragma unroll
    for (int dy = 0; dy < 2; ++dy) {
        #pragma unroll
        for (int dx = 0; dx < 2; ++dx) {
            int ix = x0i + dx, iy = y0i + dy;
            float w = (dx ? wx1 : wx0) * (dy ? wy1 : wy0);
            if (ix >= 0 && ix < Wq && iy >= 0 && iy < Hq)
                atomicAdd(&base[iy * Wq + ix], val * w);
        }
    }
}

__global__ void normalize_kernel(const float* __restrict__ acc,
                                 _Float16* __restrict__ out, int HW)
{
    int i = blockIdx.x * blockDim.x + threadIdx.x;
    if (i >= 128 * HW) return;
    int p = i & (HW - 1);   // HW is a power of two (16384)
    out[i] = (_Float16)(acc[i] / (acc[(size_t)128 * HW + p] + 1e-7f));
}

// ---------------------------------------------------------------------------
// Host-side pipeline
// ---------------------------------------------------------------------------

extern "C" void kernel_launch(void* const* d_in, const int* in_sizes, int n_in,
                              void* d_out, int out_size, void* d_ws, size_t ws_size,
                              hipStream_t stream)
{
    (void)in_sizes; (void)n_in; (void)out_size; (void)ws_size;

    const float* x    = (const float*)d_in[0];   // [4,3,512,512]
    const float* flow = (const float*)d_in[1];   // [4,2,128,128]
    const float *pb[5], *eb[4], *zb[4];
    const float* wsrc[13];
    for (int i = 0; i < 5; ++i) { wsrc[i]     = (const float*)d_in[2  + 2*i]; pb[i] = (const float*)d_in[3  + 2*i]; }
    for (int i = 0; i < 4; ++i) { wsrc[5 + i] = (const float*)d_in[12 + 2*i]; eb[i] = (const float*)d_in[13 + 2*i]; }
    for (int i = 0; i < 4; ++i) { wsrc[9 + i] = (const float*)d_in[20 + 2*i]; zb[i] = (const float*)d_in[21 + 2*i]; }

    float* out = (float*)d_out;

    // Workspace layout (bytes):
    //   A16  @ 0         : 16,777,216  (max 32*512*512 f16)
    //   B16  @ 16777216  :  8,388,608  (max 64*256*256 f16)
    //   Cacc @ 25165824  :  8,454,144  (129*16384 f32 splat accumulator)
    //   W16  @ 33619968  : ~16.4 MB    (all weights f16, rows padded to Kp)
    char* ws = (char*)d_ws;
    _Float16* A16  = (_Float16*)(ws);
    _Float16* B16  = (_Float16*)(ws + 16777216);
    float*    Cacc = (float*)   (ws + 25165824);
    _Float16* Wbase= (_Float16*)(ws + 33619968);

    // per-layer geometry: 5 pre, 4 extractor, 4 zero convs
    const int OS[13] = {32, 64, 64, 128, 128, 192, 256, 384, 512, 192, 256, 384, 512};
    const int CS[13] = { 3, 32, 64,  64, 128, 128, 192, 256, 384, 192, 256, 384, 512};

    _Float16* w16[13];
    {
        size_t off = 0;
        for (int i = 0; i < 13; ++i) {
            int K  = CS[i] * 9;
            int Kp = (K + 31) & ~31;
            int total = OS[i] * Kp;
            w16[i] = Wbase + off;
            off += (size_t)total;
            cvt_weight_kernel<<<(total + 255) / 256, 256, 0, stream>>>(
                wsrc[i], w16[i], K, Kp, total);
        }
    }

    auto conv = [&](int li, const _Float16* in, const float* bias,
                    _Float16* oh, float* of, int H, int W, int s, int act) {
        int Ci = CS[li], O = OS[li];
        int Ho = H / s, Wo = W / s;
        int Kp = (Ci * 9 + 31) & ~31;
        dim3 grid((unsigned)((Ho * Wo) / TILE_N), (unsigned)(O / TILE_M));
        conv3x3_wmma_kernel<<<grid, CONV_BLOCK, 0, stream>>>(
            in, w16[li], bias, oh, of, Ci, H, W, O, s, Ho, Wo, Kp, act);
    };

    const int HW = 128 * 128;

    for (int b = 0; b < 4; ++b) {
        // input fp32 -> f16 (into B16: 3*512*512 = 786,432 elems)
        const float* xb = x + (size_t)b * 3 * 512 * 512;
        cvt_act_kernel<<<(786432 + 255) / 256, 256, 0, stream>>>(xb, B16, 786432);

        // pre-extractor
        conv(0, B16, pb[0], A16, nullptr, 512, 512, 1, 1);   //  3->32  @512
        conv(1, A16, pb[1], B16, nullptr, 512, 512, 2, 1);   // 32->64  @256
        conv(2, B16, pb[2], A16, nullptr, 256, 256, 1, 1);   // 64->64  @256
        conv(3, A16, pb[3], B16, nullptr, 256, 256, 2, 1);   // 64->128 @128
        conv(4, B16, pb[4], A16, nullptr, 128, 128, 1, 1);   //128->128 @128 = h

        // softmax splat: A16 (feat) -> B16 (warped)
        zero_kernel<<<(129 * HW + 255) / 256, 256, 0, stream>>>(Cacc, 129 * HW);
        dim3 sg((unsigned)(HW / 256), 129u);
        splat_kernel<<<sg, 256, 0, stream>>>(A16, flow + (size_t)b * 2 * HW,
                                             Cacc, 128, 128);
        normalize_kernel<<<(128 * HW + 255) / 256, 256, 0, stream>>>(Cacc, B16, HW);

        // pyramid extractors (SiLU, f16 out) + zero convs (bias only, f32 out)
        conv(5, B16, eb[0], A16, nullptr, 128, 128, 2, 1);                       // ->192 @64
        conv(9, A16, zb[0], nullptr, out + (size_t)b * 192 * 64 * 64,
             64, 64, 1, 0);

        conv(6, A16, eb[1], B16, nullptr, 64, 64, 2, 1);                         // ->256 @32
        conv(10, B16, zb[1], nullptr, out + 3145728 + (size_t)b * 256 * 32 * 32,
             32, 32, 1, 0);

        conv(7, B16, eb[2], A16, nullptr, 32, 32, 2, 1);                         // ->384 @16
        conv(11, A16, zb[2], nullptr, out + 4194304 + (size_t)b * 384 * 16 * 16,
             16, 16, 1, 0);

        conv(8, A16, eb[3], B16, nullptr, 16, 16, 2, 1);                         // ->512 @8
        conv(12, B16, zb[3], nullptr, out + 4587520 + (size_t)b * 512 * 8 * 8,
             8, 8, 1, 0);
    }
}
